// BehaviorEncoder_3659312136834
// MI455X (gfx1250) — matrix operand order repaired and checked by reading, not compiled, not run on hardware
//
#include <hip/hip_runtime.h>
#include <hip/hip_bf16.h>

typedef __attribute__((ext_vector_type(16))) _Float16 v16h;
typedef __attribute__((ext_vector_type(8)))  _Float16 v8h;
typedef __attribute__((ext_vector_type(8)))  float    v8f;
typedef __attribute__((ext_vector_type(4)))  float    v4f;

#define T_LEN 512
#define I_DIM 32
#define H_DIM 64
#define G3    192   // 3*H
#define BT    16    // batch rows per wave (== WMMA M)

__device__ __forceinline__ v8f wmma32(v16h a, v16h b, v8f c) {
  // D(16x16 f32) = A(16x32 f16) * B(32x16 f16) + C
  return __builtin_amdgcn_wmma_f32_16x16x32_f16(false, a, false, b, (short)0, c,
                                                false, false);
}

__device__ __forceinline__ v8f splat8(float x) {
  v8f r;
#pragma unroll
  for (int i = 0; i < 8; ++i) r[i] = x;
  return r;
}

// Build a v16h A/B fragment from two 16-byte LDS chunks (ds_load_b128 each).
__device__ __forceinline__ v16h ld_frag(const _Float16* p0, const _Float16* p1) {
  union { v16h v; struct { v8h lo; v8h hi; } s; } u;
  u.s.lo = *(const v8h*)p0;
  u.s.hi = *(const v8h*)p1;
  return u.v;
}

// Fast gate math: hardware v_rcp_f32 / v_exp_f32 instead of IEEE divide chains.
__device__ __forceinline__ float sigf(float x) {
  // sigmoid(x) = rcp(1 + exp(-x))
  return __builtin_amdgcn_rcpf(1.0f + __expf(-x));
}

__device__ __forceinline__ float tanhfast(float x) {
  // tanh(x) = (1 - e) * rcp(1 + e), e = exp(-2x)
  float e = __expf(-2.0f * x);
  return (1.0f - e) * __builtin_amdgcn_rcpf(1.0f + e);
}

__global__ void __launch_bounds__(32, 1)
gru_scan_kernel(const float* __restrict__ seq,   // (B, T, I)
                const float* __restrict__ Wih,   // (3H, I)
                const float* __restrict__ Whh,   // (3H, H)
                const float* __restrict__ bih,   // (3H)
                const float* __restrict__ bhh,   // (3H)
                float* __restrict__ out)         // (B, H)
{
  __shared__ _Float16 sWih[G3 * I_DIM];    // 12 KB, f16 weights
  __shared__ _Float16 sWhh[G3 * H_DIM];    // 24 KB, f16 weights
  __shared__ _Float16 sH[2][BT * H_DIM];   // ping-pong f16 hidden tile, 4 KB

  const int lane = threadIdx.x & 31;
  const int g    = lane >> 4;   // lane group
  const int n16  = lane & 15;   // N index in a 16-wide tile / M for A-frag

  // ---- one-time: stage weights into LDS converted to f16 ----
#pragma unroll 4
  for (int i = lane; i < G3 * I_DIM; i += 32) sWih[i] = (_Float16)Wih[i];
#pragma unroll 4
  for (int i = lane; i < G3 * H_DIM; i += 32) sWhh[i] = (_Float16)Whh[i];
  for (int i = lane; i < BT * H_DIM; i += 32) sH[0][i] = (_Float16)0.0f;
  __syncthreads();

  // ---- per-lane biases (depend only on tile column N) ----
  float brz[4], bz_[4], bxn[4], bhn[4];
#pragma unroll
  for (int jt = 0; jt < 4; ++jt) {
    const int c = 16 * jt + n16;
    brz[jt] = bih[c] + bhh[c];
    bz_[jt] = bih[64 + c] + bhh[64 + c];
    bxn[jt] = bih[128 + c];
    bhn[jt] = bhh[128 + c];
  }

  // ---- preload all weight B-fragments into registers (loop-invariant) ----
  v16h wf_hh[12][2];  // [nt][kb]: K = 64 hidden -> 2 blocks of 32
  v16h wf_ih[12];     // K = 32 input  -> 1 block
#pragma unroll
  for (int nt = 0; nt < 12; ++nt) {
    const _Float16* rh = sWhh + (nt * 16 + n16) * H_DIM + 8 * g;
    wf_hh[nt][0] = ld_frag(rh, rh + 16);
    wf_hh[nt][1] = ld_frag(rh + 32, rh + 48);
    const _Float16* ri = sWih + (nt * 16 + n16) * I_DIM + 8 * g;
    wf_ih[nt] = ld_frag(ri, ri + 16);
  }

  const long b0 = (long)blockIdx.x * BT;
  const float* xrow = seq + (b0 + n16) * (long)T_LEN * I_DIM;

  // hidden state in f32, C-fragment layout: h[jt][v] = h[M=v+8g][16*jt+n16]
  v8f h[4];
#pragma unroll
  for (int jt = 0; jt < 4; ++jt) h[jt] = splat8(0.0f);

  for (int t = 0; t < T_LEN; ++t) {
    const int p = t & 1;
    const _Float16* hcur = &sH[p][0];
    _Float16* hnxt = &sH[p ^ 1][0];

    // prefetch seq a few steps ahead (global_prefetch_b8)
    if (t + 16 < T_LEN) __builtin_prefetch(xrow + (long)(t + 16) * I_DIM, 0, 0);

    // ---- x A-fragment: load f32, pack to f16 (K = 32) ----
    const float* xr = xrow + (long)t * I_DIM;
    v4f f0 = *(const v4f*)(xr + 8 * g);
    v4f f1 = *(const v4f*)(xr + 8 * g + 4);
    v4f f2 = *(const v4f*)(xr + 16 + 8 * g);
    v4f f3 = *(const v4f*)(xr + 20 + 8 * g);
    v16h ax;
#pragma unroll
    for (int i = 0; i < 4; ++i) {
      ax[i]      = (_Float16)f0[i];
      ax[4 + i]  = (_Float16)f1[i];
      ax[8 + i]  = (_Float16)f2[i];
      ax[12 + i] = (_Float16)f3[i];
    }

    // ---- h A-fragments from f16 LDS (2 x ds_load_b128 each) ----
    const _Float16* hb = hcur + n16 * H_DIM;
    v16h ah0 = ld_frag(hb + 8 * g,      hb + 16 + 8 * g);   // K = 0..31
    v16h ah1 = ld_frag(hb + 32 + 8 * g, hb + 48 + 8 * g);   // K = 32..63

#pragma unroll
    for (int jt = 0; jt < 4; ++jt) {
      v8f accr = splat8(brz[jt]);
      accr = wmma32(ah0, wf_hh[jt][0], accr);
      accr = wmma32(ah1, wf_hh[jt][1], accr);
      accr = wmma32(ax, wf_ih[jt], accr);

      v8f accz = splat8(bz_[jt]);
      accz = wmma32(ah0, wf_hh[jt + 4][0], accz);
      accz = wmma32(ah1, wf_hh[jt + 4][1], accz);
      accz = wmma32(ax, wf_ih[jt + 4], accz);

      v8f acchn = splat8(bhn[jt]);
      acchn = wmma32(ah0, wf_hh[jt + 8][0], acchn);
      acchn = wmma32(ah1, wf_hh[jt + 8][1], acchn);

      v8f accxn = splat8(bxn[jt]);
      accxn = wmma32(ax, wf_ih[jt + 8], accxn);

      // gate math in f32, elementwise in C-fragment layout
#pragma unroll
      for (int v = 0; v < 8; ++v) {
        const float r = sigf(accr[v]);
        const float z = sigf(accz[v]);
        const float n = tanhfast(accxn[v] + r * acchn[v]);
        const float hv = n + z * (h[jt][v] - n);   // (1-z)*n + z*h
        h[jt][v] = hv;
        // mirror f16 copy for next step's A-fragment: row M=v+8g, col 16*jt+n16
        hnxt[(v + 8 * g) * H_DIM + 16 * jt + n16] = (_Float16)hv;
      }
    }
    // single wave per block + in-order DS within a wave => no barrier needed
  }

  // ---- final hidden state (B, H), f32 ----
#pragma unroll
  for (int jt = 0; jt < 4; ++jt) {
#pragma unroll
    for (int v = 0; v < 8; ++v) {
      out[(b0 + v + 8 * g) * H_DIM + 16 * jt + n16] = h[jt][v];
    }
  }
}

extern "C" void kernel_launch(void* const* d_in, const int* in_sizes, int n_in,
                              void* d_out, int out_size, void* d_ws, size_t ws_size,
                              hipStream_t stream) {
  const float* seq = (const float*)d_in[0];   // (4096, 512, 32)
  const float* Wih = (const float*)d_in[1];   // (192, 32)
  const float* Whh = (const float*)d_in[2];   // (192, 64)
  const float* bih = (const float*)d_in[3];   // (192)
  const float* bhh = (const float*)d_in[4];   // (192)
  float* out = (float*)d_out;                 // (4096, 64)

  const int B = 4096;
  dim3 grid(B / BT);   // 256 blocks, one wave32 each -> 256 independent scans
  dim3 block(32);
  gru_scan_kernel<<<grid, block, 0, stream>>>(seq, Wih, Whh, bih, bhh, out);
}
